// AttentionAutoEncoder_47270410060239
// MI455X (gfx1250) — matrix operand order repaired and checked by reading, not compiled, not run on hardware
//
#include <hip/hip_runtime.h>
#include <hip/hip_bf16.h>
#include <math.h>

// ---------------------------------------------------------------------------
// AttentionAutoEncoder on gfx1250 (MI455X):
//   Q = Wq@X ; K = Wv@X ; V = Wk@X       (reference swaps K/V weights)
//   S = (Q@K^T)/sqrt(D) ; Z = softmax(S) ; A = Z@V
// Compute-bound (~344 GFLOP vs ~30us of HBM traffic at 23.3 TB/s) -> all five
// GEMMs run through v_wmma_f32_16x16x32_bf16 with f32 accumulation.
// X is transposed once (memory-bound, ~3us) and V is produced transposed by
// the projection epilogue, so every GEMM stages B as [n][k] with pure b128
// traffic through GLOBAL_LOAD_ASYNC_TO_LDS_B128 (ASYNCcnt-tracked).
// ---------------------------------------------------------------------------

typedef __bf16 bf16_t;
typedef __attribute__((ext_vector_type(16))) __bf16 v16bf;
typedef __attribute__((ext_vector_type(8)))  __bf16 v8bf;
typedef __attribute__((ext_vector_type(8)))  float  v8f;

#define D_DIM 2048
#define N_DIM 8192

#if defined(__HIP_DEVICE_COMPILE__) && __has_builtin(__builtin_amdgcn_global_load_async_to_lds_b128)
#define HAVE_ASYNC_LDS 1
// Builtin signature (from clang diagnostic): param0 = v4i in AS(1) [global],
// param1 = v4i in AS(3) [LDS], then imm offset, imm cpol.
typedef int v4i_t __attribute__((vector_size(16)));
typedef __attribute__((address_space(1))) v4i_t* gas_ptr_t;
typedef __attribute__((address_space(3))) v4i_t* lds_ptr_t;
#else
#define HAVE_ASYNC_LDS 0
#endif

__device__ __forceinline__ void wait_async_zero() {
#if HAVE_ASYNC_LDS
#if __has_builtin(__builtin_amdgcn_s_wait_asynccnt)
  __builtin_amdgcn_s_wait_asynccnt(0);
#else
  asm volatile("s_wait_asynccnt 0x0" ::: "memory");
#endif
#endif
}

// ---------------------------------------------------------------- convert ---
__global__ __launch_bounds__(256) void f32_to_bf16_kernel(
    const float* __restrict__ in, bf16_t* __restrict__ out, int n) {
  int i = blockIdx.x * blockDim.x + threadIdx.x;
  int stride = gridDim.x * blockDim.x;
  for (; i < n; i += stride) out[i] = (bf16_t)in[i];
}

// -------------------------------------------------- transpose + downconvert -
// in: (R x C) f32 row-major  ->  out: (C x R) bf16 row-major. 64x64 LDS tiles.
__global__ __launch_bounds__(256) void transpose_f32_to_bf16(
    const float* __restrict__ in, bf16_t* __restrict__ out, int R, int C) {
  __shared__ bf16_t tile[64][72];              // padded to dodge bank conflicts
  const int r0  = blockIdx.y * 64;
  const int c0  = blockIdx.x * 64;
  const int tr  = threadIdx.x >> 4;            // 0..15
  const int tc4 = (threadIdx.x & 15) * 4;      // 0,4,...,60
#pragma unroll
  for (int it = 0; it < 4; ++it) {
    const int row = it * 16 + tr;
    const float4 v = *(const float4*)&in[(size_t)(r0 + row) * C + c0 + tc4];
    tile[row][tc4 + 0] = (bf16_t)v.x;
    tile[row][tc4 + 1] = (bf16_t)v.y;
    tile[row][tc4 + 2] = (bf16_t)v.z;
    tile[row][tc4 + 3] = (bf16_t)v.w;
  }
  __syncthreads();
#pragma unroll
  for (int it = 0; it < 4; ++it) {
    const int orow = it * 16 + tr;             // output row = original column
    union { bf16_t e[4]; uint2 u; } p;
    p.e[0] = tile[tc4 + 0][orow];
    p.e[1] = tile[tc4 + 1][orow];
    p.e[2] = tile[tc4 + 2][orow];
    p.e[3] = tile[tc4 + 3][orow];
    *(uint2*)&out[(size_t)(c0 + orow) * R + r0 + tc4] = p.u;
  }
}

// ------------------------------------------------------------------- GEMM ---
// C(MxN) = A(MxK) * B ;  A row-major (lda) ; B supplied TRANSPOSED: Bt is
// (N x K) row-major (ldb) == the LDS [n][k] layout, so staging is pure b128.
// OUT_MODE: 0 = f32 row-major, 1 = bf16 row-major, 2 = bf16 transposed (C^T,
//           ldc = row length of C^T = M) — packed 8-wide b128 stores.
// Block tile 128x128, 256 threads = 8 waves, wave tile 64x32 (4x2 WMMA),
// K-step 32, LDS stride 40 bf16 (80B) for conflict-free b128 fragment loads.
template <int OUT_MODE>
__global__ __launch_bounds__(256) void gemm_wmma_bf16(
    const bf16_t* __restrict__ A, const bf16_t* __restrict__ Bt,
    void* __restrict__ C, int M, int N, int K, int lda, int ldb, int ldc) {
  __shared__ bf16_t sA[128 * 40];              // [m][k]
  __shared__ bf16_t sB[128 * 40];              // [n][k]

  const int tid  = threadIdx.x;
  const int wave = tid >> 5;
  const int lane = tid & 31;
  const int half = lane >> 4;                  // 0 | 1
  const int lm   = lane & 15;                  // 0..15
  const int wr   = wave >> 2;                  // 0..1 -> M offset 64*wr
  const int wc   = wave & 3;                   // 0..3 -> N offset 32*wc

  const int m0 = blockIdx.y * 128;
  const int n0 = blockIdx.x * 128;

  v8f acc[4][2];
#pragma unroll
  for (int i = 0; i < 4; ++i)
#pragma unroll
    for (int j = 0; j < 2; ++j)
      acc[i][j] = (v8f){0.f, 0.f, 0.f, 0.f, 0.f, 0.f, 0.f, 0.f};

  const int srow = tid >> 1;                   // 0..127
  const int scol = (tid & 1) << 4;             // 0 | 16

  union V16 { v16bf v; v8bf h[2]; };

  for (int k0 = 0; k0 < K; k0 += 32) {
    const bf16_t* gA = A  + (size_t)(m0 + srow) * lda + k0 + scol;
    const bf16_t* gB = Bt + (size_t)(n0 + srow) * ldb + k0 + scol;
    bf16_t* lA = &sA[srow * 40 + scol];
    bf16_t* lB = &sB[srow * 40 + scol];
#if HAVE_ASYNC_LDS
    // CDNA5 async copy: memory -> LDS without the VGPR round-trip (ASYNCcnt).
    __builtin_amdgcn_global_load_async_to_lds_b128(
        (gas_ptr_t)(void*)gA, (lds_ptr_t)(void*)lA, 0, 0);
    __builtin_amdgcn_global_load_async_to_lds_b128(
        (gas_ptr_t)(void*)(gA + 8), (lds_ptr_t)(void*)(lA + 8), 0, 0);
    __builtin_amdgcn_global_load_async_to_lds_b128(
        (gas_ptr_t)(void*)gB, (lds_ptr_t)(void*)lB, 0, 0);
    __builtin_amdgcn_global_load_async_to_lds_b128(
        (gas_ptr_t)(void*)(gB + 8), (lds_ptr_t)(void*)(lB + 8), 0, 0);
#else
    {
      uint4 a0 = *(const uint4*)gA;
      uint4 a1 = *(const uint4*)(gA + 8);
      uint4 b0 = *(const uint4*)gB;
      uint4 b1 = *(const uint4*)(gB + 8);
      *(uint4*)lA       = a0;
      *(uint4*)(lA + 8) = a1;
      *(uint4*)lB       = b0;
      *(uint4*)(lB + 8) = b1;
    }
#endif
    if (k0 + 32 < K) {                         // warm L2 for the next K tile
      __builtin_prefetch(gA + 32, 0, 1);
      __builtin_prefetch(gB + 32, 0, 1);
    }
    wait_async_zero();
    __syncthreads();

    // ---- fragments (contiguous k order, same for A and B) ----
    // B 32x16: lane holds column n=lm, K = half*16 + i (i=0..15)
    v16bf bfrag[2];
#pragma unroll
    for (int j = 0; j < 2; ++j) {
      const bf16_t* p = &sB[(wc * 32 + j * 16 + lm) * 40 + half * 16];
      V16 t; t.h[0] = *(const v8bf*)p; t.h[1] = *(const v8bf*)(p + 8);
      bfrag[j] = t.v;
    }
    // A 16x32: lane holds row m=lm, K = half*16 + i (i=0..15)
    v16bf afrag[4];
#pragma unroll
    for (int i = 0; i < 4; ++i) {
      const bf16_t* p = &sA[(wr * 64 + i * 16 + lm) * 40];
      V16 t;
      t.h[0] = *(const v8bf*)(p + half * 16);
      t.h[1] = *(const v8bf*)(p + half * 16 + 8);
      afrag[i] = t.v;
    }

#pragma unroll
    for (int i = 0; i < 4; ++i)
#pragma unroll
      for (int j = 0; j < 2; ++j)
        acc[i][j] = __builtin_amdgcn_wmma_f32_16x16x32_bf16(
            false, afrag[i], false, bfrag[j], (short)0, acc[i][j], false, false);

    __syncthreads();
  }

  // ---- epilogue: C/D layout -> VGPR r holds row (r + 8*half), col lm ----
#pragma unroll
  for (int i = 0; i < 4; ++i)
#pragma unroll
    for (int j = 0; j < 2; ++j) {
      if (OUT_MODE == 2) {
        // transposed bf16 out: the 8 accumulator rows are contiguous in C^T
        const int ct_row = n0 + wc * 32 + j * 16 + lm;
        const int base   = m0 + wr * 64 + i * 16 + half * 8;
        union { bf16_t e[8]; uint4 u; } p;
#pragma unroll
        for (int r = 0; r < 8; ++r) p.e[r] = (bf16_t)acc[i][j][r];
        *(uint4*)&((bf16_t*)C)[(size_t)ct_row * ldc + base] = p.u;
      } else {
#pragma unroll
        for (int r = 0; r < 8; ++r) {
          const int row = m0 + wr * 64 + i * 16 + half * 8 + r;
          const int col = n0 + wc * 32 + j * 16 + lm;
          const float v = acc[i][j][r];
          if (OUT_MODE == 1) ((bf16_t*)C)[(size_t)row * ldc + col] = (bf16_t)v;
          else               ((float*) C)[(size_t)row * ldc + col] = v;
        }
      }
    }
}

// ---------------------------------------------------------------- softmax ---
// One block per row (2048 cols, 8 per thread); 1/sqrt(D) folded in here.
__global__ __launch_bounds__(256) void softmax_bf16_kernel(
    const float* __restrict__ S, bf16_t* __restrict__ Z, float scale) {
  __shared__ float red[256];
  const int row = blockIdx.x;
  const float* s = S + (size_t)row * D_DIM;

  float vals[8];
  float m = -INFINITY;
#pragma unroll
  for (int i = 0; i < 8; ++i) {
    vals[i] = s[threadIdx.x + i * 256];
    m = fmaxf(m, vals[i]);
  }
  red[threadIdx.x] = m;
  __syncthreads();
  for (int off = 128; off > 0; off >>= 1) {
    if (threadIdx.x < off)
      red[threadIdx.x] = fmaxf(red[threadIdx.x], red[threadIdx.x + off]);
    __syncthreads();
  }
  m = red[0];
  __syncthreads();

  float sum = 0.f;
#pragma unroll
  for (int i = 0; i < 8; ++i) {
    vals[i] = __expf(scale * (vals[i] - m));
    sum += vals[i];
  }
  red[threadIdx.x] = sum;
  __syncthreads();
  for (int off = 128; off > 0; off >>= 1) {
    if (threadIdx.x < off) red[threadIdx.x] += red[threadIdx.x + off];
    __syncthreads();
  }
  const float inv = 1.0f / red[0];

  bf16_t* z = Z + (size_t)row * D_DIM;
#pragma unroll
  for (int i = 0; i < 8; ++i)
    z[threadIdx.x + i * 256] = (bf16_t)(vals[i] * inv);
}

// ----------------------------------------------------------------- launch ---
extern "C" void kernel_launch(void* const* d_in, const int* in_sizes, int n_in,
                              void* d_out, int out_size, void* d_ws, size_t ws_size,
                              hipStream_t stream) {
  const float* X  = (const float*)d_in[0];   // (D, N)
  const float* Wq = (const float*)d_in[1];   // (D, D)
  const float* Wk = (const float*)d_in[2];   // (D, D)
  const float* Wv = (const float*)d_in[3];   // (D, D)
  float* out = (float*)d_out;                // (D, N) f32

  const int D = D_DIM, N = N_DIM;
  char* ws = (char*)d_ws;
  size_t off = 0;
  auto alloc = [&](size_t bytes) -> char* {
    char* p = ws + off;
    off += (bytes + 255) & ~(size_t)255;
    return p;
  };
  bf16_t* XbT = (bf16_t*)alloc((size_t)N * D * 2);   // X^T  (N x D), 32 MB
  bf16_t* Wqb = (bf16_t*)alloc((size_t)D * D * 2);   //  8 MB
  bf16_t* Wkb = (bf16_t*)alloc((size_t)D * D * 2);
  bf16_t* Wvb = (bf16_t*)alloc((size_t)D * D * 2);
  bf16_t* Qb  = (bf16_t*)alloc((size_t)D * N * 2);   // Q    (D x N)
  bf16_t* Kb  = (bf16_t*)alloc((size_t)D * N * 2);   // K    (D x N)
  bf16_t* VbT = (bf16_t*)alloc((size_t)N * D * 2);   // V^T  (N x D)
  float*  Sm  = (float*) alloc((size_t)D * D * 4);   // 16 MB
  bf16_t* Zb  = (bf16_t*)alloc((size_t)D * D * 2);

  // 1) downconvert weights; transpose+downconvert X (one-time, memory-bound)
  f32_to_bf16_kernel<<<1024, 256, 0, stream>>>(Wq, Wqb, D * D);
  f32_to_bf16_kernel<<<1024, 256, 0, stream>>>(Wk, Wkb, D * D);
  f32_to_bf16_kernel<<<1024, 256, 0, stream>>>(Wv, Wvb, D * D);
  transpose_f32_to_bf16<<<dim3(N / 64, D / 64), 256, 0, stream>>>(X, XbT, D, N);

  const dim3 blk(256);
  const dim3 gProj(N / 128, D / 128);        // (64, 16)
  // 2) projections (reference swap: K uses W_v, V uses W_k); V emitted as V^T
  gemm_wmma_bf16<1><<<gProj, blk, 0, stream>>>(Wqb, XbT, Qb,  D, N, D, D, D, N);
  gemm_wmma_bf16<1><<<gProj, blk, 0, stream>>>(Wvb, XbT, Kb,  D, N, D, D, D, N);
  gemm_wmma_bf16<2><<<gProj, blk, 0, stream>>>(Wkb, XbT, VbT, D, N, D, D, D, D);

  // 3) scores = Q @ K^T : K row-major (D x N) is already the Bt layout
  const dim3 gScore(D / 128, D / 128);       // (16, 16)
  gemm_wmma_bf16<0><<<gScore, blk, 0, stream>>>(Qb, Kb, Sm, D, D, N, N, N, D);

  // 4) row softmax with 1/sqrt(D) folded in, emit bf16 Z
  const float scale = 1.0f / sqrtf((float)D);
  softmax_bf16_kernel<<<D, 256, 0, stream>>>(Sm, Zb, scale);

  // 5) A = Z @ V = Z @ (V^T)^T -> f32 output
  gemm_wmma_bf16<0><<<gProj, blk, 0, stream>>>(Zb, VbT, out, D, N, D, D, D, N);
}
